// DecoderLayer_3186865734405
// MI455X (gfx1250) — matrix operand order repaired and checked by reading, not compiled, not run on hardware
//
#include <hip/hip_runtime.h>
#include <hip/hip_bf16.h>

typedef __attribute__((ext_vector_type(16))) __bf16 v16bf;
typedef __attribute__((ext_vector_type(8)))  float  v8f;

#define NEGBIAS (-1e9f)

// CDNA5 async global->LDS DMA (ASYNCcnt-tracked), per-lane 16B.
#define ASYNC_LOAD_B128(lds_ptr, gptr)                                         \
    asm volatile("global_load_async_to_lds_b128 %0, %1, off"                   \
                 :: "v"((unsigned int)(uintptr_t)(lds_ptr)),                   \
                    "v"((unsigned long long)(uintptr_t)(gptr))                 \
                 : "memory")
#define WAIT_ASYNC(n) asm volatile("s_wait_asynccnt %0" :: "i"(n) : "memory")

// ---------------------------------------------------------------------------
// Fragment load helpers (CDNA5 WMMA bf16 16x16x32 layouts)
// load_frag_a (unused, AMD table order): element e -> K = ((e&8)<<1) + lh*8 + (e&7)  => two 16B chunks
// A and B fragments as loaded here (load_frag_contig): element e -> K = lh*16 + e                  => one 32B chunk
// ---------------------------------------------------------------------------
__device__ __forceinline__ v16bf load_frag_contig(const __bf16* p) {
    union { v16bf v; uint4 u[2]; } r;
    r.u[0] = *(const uint4*)(p);
    r.u[1] = *(const uint4*)(p + 8);
    return r.v;
}
__device__ __forceinline__ v16bf load_frag_a(const __bf16* rowp, int lh) {
    union { v16bf v; uint4 u[2]; } r;
    r.u[0] = *(const uint4*)(rowp + (lh << 3));
    r.u[1] = *(const uint4*)(rowp + 16 + (lh << 3));
    return r.v;
}

// ---------------------------------------------------------------------------
// f32 -> bf16 conversion (weights), 4 elements/thread
// ---------------------------------------------------------------------------
__global__ __launch_bounds__(256)
void cvt_f32_bf16_kernel(const float* __restrict__ in, __bf16* __restrict__ out, int n4) {
    int i = blockIdx.x * 256 + threadIdx.x;
    if (i < n4) {
        float4 v = ((const float4*)in)[i];
        union { uint2 u; __bf16 h[4]; } w;
        w.h[0] = (__bf16)v.x; w.h[1] = (__bf16)v.y;
        w.h[2] = (__bf16)v.z; w.h[3] = (__bf16)v.w;
        ((uint2*)out)[i] = w.u;
    }
}

// ---------------------------------------------------------------------------
// LayerNorm over C=512, one row per block, bf16 output
// ---------------------------------------------------------------------------
__global__ __launch_bounds__(256)
void layernorm_bf16_kernel(const float* __restrict__ x, const float* __restrict__ g,
                           const float* __restrict__ b, __bf16* __restrict__ out, int C) {
    __shared__ float r1[256];
    __shared__ float r2[256];
    const int row = blockIdx.x;
    const float* xr = x + (size_t)row * C;
    const int c0 = threadIdx.x * 2;            // C == 512: one float2 per thread
    float2 v = *(const float2*)(xr + c0);
    r1[threadIdx.x] = v.x + v.y;
    r2[threadIdx.x] = v.x * v.x + v.y * v.y;
    __syncthreads();
    for (int off = 128; off; off >>= 1) {
        if (threadIdx.x < off) { r1[threadIdx.x] += r1[threadIdx.x + off];
                                 r2[threadIdx.x] += r2[threadIdx.x + off]; }
        __syncthreads();
    }
    const float mean = r1[0] / C;
    const float var  = r2[0] / C - mean * mean;
    const float rstd = rsqrtf(var + 1e-5f);
    union { unsigned int u; __bf16 h[2]; } w;
    w.h[0] = (__bf16)((v.x - mean) * rstd * g[c0] + b[c0]);
    w.h[1] = (__bf16)((v.y - mean) * rstd * g[c0 + 1] + b[c0 + 1]);
    *(unsigned int*)(out + (size_t)row * C + c0) = w.u;
}

// ---------------------------------------------------------------------------
// Per-(b,q) "any allowed key" bit for the empty-mask fallback
// ---------------------------------------------------------------------------
__global__ __launch_bounds__(256)
void rowany_kernel(const int* __restrict__ mask, int* __restrict__ out, int Mkv) {
    __shared__ int red[256];
    const int row = blockIdx.x;
    const int* mr = mask + (size_t)row * Mkv;
    int any = 0;
    for (int c = threadIdx.x; c < Mkv; c += 256) any |= (mr[c] != 0);
    red[threadIdx.x] = any;
    __syncthreads();
    for (int off = 128; off; off >>= 1) {
        if (threadIdx.x < off) red[threadIdx.x] |= red[threadIdx.x + off];
        __syncthreads();
    }
    if (threadIdx.x == 0) out[row] = red[0];
}

// ---------------------------------------------------------------------------
// bf16 WMMA GEMM: out[M,N] = A[M,K]*B[K,N] (+bias +residual, opt ReLU)
// Block = 256 threads = 8 waves, tile 64(M) x 64(N), K-step 32.
// A tiles streamed with double-buffered global_load_async_to_lds_b128.
// ---------------------------------------------------------------------------
__global__ __launch_bounds__(256)
void gemm_bf16_kernel(const __bf16* __restrict__ A, int lda,
                      const __bf16* __restrict__ Bm, int ldb,
                      const float* __restrict__ bias,
                      const float* __restrict__ resid, int ldr,
                      float* __restrict__ outF, __bf16* __restrict__ outB, int ldc,
                      int M, int N, int K, int doRelu) {
    __shared__ __bf16 As[2][64][32];   // [buf][m][k]
    __shared__ __bf16 Bs[64][32];      // [n][k] (transposed -> contiguous B-frags)
    const int tid = threadIdx.x;
    const int lane = tid & 31, wave = tid >> 5;
    const int lh = lane >> 4, l15 = lane & 15;
    const int rw = wave >> 1, cw = wave & 1;
    const int m0 = blockIdx.y * 64, n0 = blockIdx.x * 64;

    // staging coords (row-clamped: OOB rows produce discarded output rows)
    const int arow = tid >> 2;                 // 0..63
    const int acol = (tid & 3) << 3;           // 0,8,16,24
    const int arowc = (m0 + arow < M) ? (m0 + arow) : (M - 1);
    const __bf16* asrc = A + (size_t)arowc * lda + acol;
    const int bk  = tid >> 3;                  // 0..31
    const int bn8 = (tid & 7) << 3;            // 0..56

    const int nT = K >> 5;
    ASYNC_LOAD_B128(&As[0][arow][acol], asrc);          // prologue: tile 0

    v8f acc0 = {}, acc1 = {};
    for (int t = 0; t < nT; ++t) {
        const int k0 = t << 5;
        if (t + 1 < nT) {                               // prefetch tile t+1
            ASYNC_LOAD_B128(&As[(t + 1) & 1][arow][acol], asrc + k0 + 32);
            WAIT_ASYNC(1);                              // tile t landed
        } else {
            WAIT_ASYNC(0);
        }
        {   // B tile 32(K)x64(N): vector read, transposed scatter to LDS
            union { uint4 u; __bf16 h[8]; } w;
            w.u = *(const uint4*)(Bm + (size_t)(k0 + bk) * ldb + n0 + bn8);
            #pragma unroll
            for (int j = 0; j < 8; ++j) Bs[bn8 + j][bk] = w.h[j];
        }
        __syncthreads();
        const __bf16* ar = &As[t & 1][rw * 16 + l15][0];
        const v16bf af = load_frag_contig(ar + (lh << 4));
        const v16bf b0 = load_frag_contig(&Bs[cw * 32 + l15][lh << 4]);
        const v16bf b1 = load_frag_contig(&Bs[cw * 32 + 16 + l15][lh << 4]);
        acc0 = __builtin_amdgcn_wmma_f32_16x16x32_bf16(false, af, false, b0, (short)0, acc0, false, false);
        acc1 = __builtin_amdgcn_wmma_f32_16x16x32_bf16(false, af, false, b1, (short)0, acc1, false, false);
        __syncthreads();
    }
    #pragma unroll
    for (int r = 0; r < 8; ++r) {
        const int m = m0 + rw * 16 + r + (lh << 3);
        if (m < M) {
            #pragma unroll
            for (int cc = 0; cc < 2; ++cc) {
                const int n = n0 + cw * 32 + cc * 16 + l15;
                float v = (cc == 0) ? acc0[r] : acc1[r];
                if (bias)  v += bias[n];
                if (resid) v += resid[(size_t)m * ldr + n];
                if (doRelu) v = fmaxf(v, 0.f);
                if (outF) outF[(size_t)m * ldc + n] = v;
                else      outB[(size_t)m * ldc + n] = (__bf16)v;
            }
        }
    }
}

// ---------------------------------------------------------------------------
// Flash attention: one wave per (b, h, 16-query tile).
// ---------------------------------------------------------------------------
__global__ __launch_bounds__(32)
void flash_attn_kernel(const __bf16* __restrict__ Qp, int ldq,
                       const __bf16* __restrict__ Kp, int ldk,
                       const __bf16* __restrict__ Vp, int ldv,
                       __bf16* __restrict__ Out, int ldo,
                       const int* __restrict__ mask, const int* __restrict__ rowany,
                       int H, int Nq, int Mkv, float scale) {
    __shared__ __bf16 pbuf[16][32];   // P tile: D-layout f32 -> A-layout bf16 relay
    __shared__ __bf16 vtr[64][32];    // V tile transposed: [dim][key]
    const int lane = threadIdx.x;
    const int lh = lane >> 4, l15 = lane & 15;
    const int b = blockIdx.x / H, h = blockIdx.x % H;
    const int qt = blockIdx.y;

    // Q fragments, row-clamped (tail rows discarded at epilogue)
    const int qm = qt * 16 + l15;
    const int qrow = b * Nq + (qm < Nq ? qm : Nq - 1);
    const __bf16* qbase = Qp + (size_t)qrow * ldq + h * 64;
    const v16bf qa0 = load_frag_contig(qbase + (lh << 4));
    const v16bf qa1 = load_frag_contig(qbase + 32 + (lh << 4));

    // per-row mask info hoisted out of the key loop
    const int* mrow[8];
    int anyq[8];
    #pragma unroll
    for (int r = 0; r < 8; ++r) {
        const int q = qt * 16 + r + (lh << 3);
        const int qc = (q < Nq ? q : Nq - 1);
        mrow[r] = mask ? (mask + (size_t)(b * Nq + qc) * Mkv) : nullptr;
        anyq[r] = (rowany && q < Nq) ? rowany[b * Nq + qc] : 0;
    }

    v8f acc0 = {}, acc1 = {}, acc2 = {}, acc3 = {};
    float m8[8], l8[8];
    #pragma unroll
    for (int r = 0; r < 8; ++r) { m8[r] = -3.0e38f; l8[r] = 0.f; }

    const int nkb = (Mkv + 31) >> 5;
    for (int kb = 0; kb < nkb; ++kb) {
        const int key0 = kb << 5;
        v8f s[2];
        #pragma unroll
        for (int t = 0; t < 2; ++t) {            // S[16q x 32k] in two halves
            const int key = key0 + t * 16 + l15;
            const int krow = b * Mkv + (key < Mkv ? key : Mkv - 1);
            const __bf16* kbase = Kp + (size_t)krow * ldk + h * 64 + (lh << 4);
            const v16bf kf0 = load_frag_contig(kbase);
            const v16bf kf1 = load_frag_contig(kbase + 32);
            v8f sv = {};
            sv = __builtin_amdgcn_wmma_f32_16x16x32_bf16(false, qa0, false, kf0, (short)0, sv, false, false);
            sv = __builtin_amdgcn_wmma_f32_16x16x32_bf16(false, qa1, false, kf1, (short)0, sv, false, false);
            s[t] = sv;
        }
        // scale + mask bias
        #pragma unroll
        for (int t = 0; t < 2; ++t) {
            const int key = key0 + t * 16 + l15;
            #pragma unroll
            for (int r = 0; r < 8; ++r) {
                float bias = 0.f;
                if (key >= Mkv) bias = NEGBIAS;
                else if (mask) {
                    if (!mrow[r][key] && anyq[r]) bias = NEGBIAS;
                }
                s[t][r] = s[t][r] * scale + bias;
            }
        }
        // online softmax (16-lane row reductions stay inside each half-wave)
        #pragma unroll
        for (int r = 0; r < 8; ++r) {
            float v = fmaxf(s[0][r], s[1][r]);
            v = fmaxf(v, __shfl_xor(v, 1, 32));
            v = fmaxf(v, __shfl_xor(v, 2, 32));
            v = fmaxf(v, __shfl_xor(v, 4, 32));
            v = fmaxf(v, __shfl_xor(v, 8, 32));
            const float newm = fmaxf(m8[r], v);
            const float alpha = __expf(m8[r] - newm);
            const float p0 = __expf(s[0][r] - newm);
            const float p1 = __expf(s[1][r] - newm);
            float rs = p0 + p1;
            rs += __shfl_xor(rs, 1, 32);
            rs += __shfl_xor(rs, 2, 32);
            rs += __shfl_xor(rs, 4, 32);
            rs += __shfl_xor(rs, 8, 32);
            l8[r] = l8[r] * alpha + rs;
            m8[r] = newm;
            acc0[r] *= alpha; acc1[r] *= alpha; acc2[r] *= alpha; acc3[r] *= alpha;
            pbuf[r + (lh << 3)][l15]      = (__bf16)p0;
            pbuf[r + (lh << 3)][16 + l15] = (__bf16)p1;
        }
        // stage V transposed: lane owns one (clamped) key row
        {
            const int key = key0 + lane;
            const int vrow = b * Mkv + (key < Mkv ? key : Mkv - 1);
            const __bf16* vsrc = Vp + (size_t)vrow * ldv + h * 64;
            #pragma unroll
            for (int c8 = 0; c8 < 8; ++c8) {
                union { uint4 u; __bf16 hh[8]; } w;
                w.u = *(const uint4*)(vsrc + c8 * 8);
                #pragma unroll
                for (int j = 0; j < 8; ++j) vtr[c8 * 8 + j][lane] = w.hh[j];
            }
        }
        __syncthreads();   // single-wave WG: cheap; orders cross-lane LDS handoff
        const v16bf pf  = load_frag_contig(&pbuf[l15][lh << 4]);
        const v16bf vf0 = load_frag_contig(&vtr[ 0 + l15][lh << 4]);
        const v16bf vf1 = load_frag_contig(&vtr[16 + l15][lh << 4]);
        const v16bf vf2 = load_frag_contig(&vtr[32 + l15][lh << 4]);
        const v16bf vf3 = load_frag_contig(&vtr[48 + l15][lh << 4]);
        acc0 = __builtin_amdgcn_wmma_f32_16x16x32_bf16(false, pf, false, vf0, (short)0, acc0, false, false);
        acc1 = __builtin_amdgcn_wmma_f32_16x16x32_bf16(false, pf, false, vf1, (short)0, acc1, false, false);
        acc2 = __builtin_amdgcn_wmma_f32_16x16x32_bf16(false, pf, false, vf2, (short)0, acc2, false, false);
        acc3 = __builtin_amdgcn_wmma_f32_16x16x32_bf16(false, pf, false, vf3, (short)0, acc3, false, false);
        __syncthreads();
    }
    #pragma unroll
    for (int r = 0; r < 8; ++r) {
        const int q = qt * 16 + r + (lh << 3);
        if (q < Nq) {
            const float inv = 1.f / l8[r];
            __bf16* o = Out + (size_t)(b * Nq + q) * ldo + h * 64;
            o[ 0 + l15] = (__bf16)(acc0[r] * inv);
            o[16 + l15] = (__bf16)(acc1[r] * inv);
            o[32 + l15] = (__bf16)(acc2[r] * inv);
            o[48 + l15] = (__bf16)(acc3[r] * inv);
        }
    }
}

// ---------------------------------------------------------------------------
// Host-side orchestration
// ---------------------------------------------------------------------------
extern "C" void kernel_launch(void* const* d_in, const int* in_sizes, int n_in,
                              void* d_out, int out_size, void* d_ws, size_t ws_size,
                              hipStream_t stream) {
    (void)in_sizes; (void)n_in; (void)out_size; (void)ws_size;
    const int B = 8, Nq = 100, HW = 4096, C = 512, F = 2048, H = 8;
    const int Mq = B * Nq;        // 800
    const int Mkv = B * HW;       // 32768
    const float scale = 0.125f;   // 1/sqrt(64)

    const float* q_in   = (const float*)d_in[0];
    const float* kv_in  = (const float*)d_in[1];
    const int*   mask   = (const int*)d_in[2];
    const float* xaWqkv = (const float*)d_in[3];
    const float* xabqkv = (const float*)d_in[4];
    const float* xaWo   = (const float*)d_in[5];
    const float* xabo   = (const float*)d_in[6];
    const float* xagq   = (const float*)d_in[7];
    const float* xabq   = (const float*)d_in[8];
    const float* xagkv  = (const float*)d_in[9];
    const float* xabkv  = (const float*)d_in[10];
    const float* saWqkv = (const float*)d_in[11];
    const float* sabqkv = (const float*)d_in[12];
    const float* saWo   = (const float*)d_in[13];
    const float* sabo   = (const float*)d_in[14];
    const float* n1g    = (const float*)d_in[15];
    const float* n1b    = (const float*)d_in[16];
    const float* n2g    = (const float*)d_in[17];
    const float* n2b    = (const float*)d_in[18];
    const float* ffW1   = (const float*)d_in[19];
    const float* ffb1   = (const float*)d_in[20];
    const float* ffW2   = (const float*)d_in[21];
    const float* ffb2   = (const float*)d_in[22];

    char* wsp = (char*)d_ws;
    auto alloc = [&](size_t bytes) -> char* {
        char* p = wsp; wsp += (bytes + 255) & ~(size_t)255; return p;
    };
    __bf16* qn     = (__bf16*)alloc((size_t)Mq * C * 2);
    __bf16* kvn    = (__bf16*)alloc((size_t)Mkv * C * 2);
    __bf16* qp     = (__bf16*)alloc((size_t)Mq * C * 2);
    __bf16* kvp    = (__bf16*)alloc((size_t)Mkv * 1024 * 2);
    __bf16* xaout  = (__bf16*)alloc((size_t)Mq * C * 2);
    float*  q1     = (float*)alloc((size_t)Mq * C * 4);
    __bf16* qn1    = (__bf16*)alloc((size_t)Mq * C * 2);
    __bf16* saqkv  = (__bf16*)alloc((size_t)Mq * 1536 * 2);
    __bf16* saout  = (__bf16*)alloc((size_t)Mq * C * 2);
    float*  q2     = (float*)alloc((size_t)Mq * C * 4);
    __bf16* qn2    = (__bf16*)alloc((size_t)Mq * C * 2);
    __bf16* hbuf   = (__bf16*)alloc((size_t)Mq * F * 2);
    __bf16* wXAqkv = (__bf16*)alloc((size_t)C * 3 * C * 2);
    __bf16* wXAo   = (__bf16*)alloc((size_t)C * C * 2);
    __bf16* wSAqkv = (__bf16*)alloc((size_t)C * 3 * C * 2);
    __bf16* wSAo   = (__bf16*)alloc((size_t)C * C * 2);
    __bf16* wF1    = (__bf16*)alloc((size_t)C * F * 2);
    __bf16* wF2    = (__bf16*)alloc((size_t)F * C * 2);
    int*    ranyb  = (int*)alloc((size_t)Mq * 4);

    auto cvt = [&](const float* s, __bf16* d, int n) {
        cvt_f32_bf16_kernel<<<(n / 4 + 255) / 256, 256, 0, stream>>>(s, d, n / 4);
    };
    cvt(xaWqkv, wXAqkv, C * 3 * C);
    cvt(xaWo,   wXAo,   C * C);
    cvt(saWqkv, wSAqkv, C * 3 * C);
    cvt(saWo,   wSAo,   C * C);
    cvt(ffW1,   wF1,    C * F);
    cvt(ffW2,   wF2,    F * C);

    layernorm_bf16_kernel<<<Mq,  256, 0, stream>>>(q_in,  xagq,  xabq,  qn,  C);
    layernorm_bf16_kernel<<<Mkv, 256, 0, stream>>>(kv_in, xagkv, xabkv, kvn, C);
    rowany_kernel<<<Mq, 256, 0, stream>>>(mask, ranyb, HW);

    const int gy_q  = (Mq + 63) / 64;     // 13
    const int gy_kv = (Mkv + 63) / 64;    // 512

    // qp = LN(q) @ Wq + bq                         [800, 512] bf16
    gemm_bf16_kernel<<<dim3(C / 64, gy_q), 256, 0, stream>>>(
        qn, C, wXAqkv, 3 * C, xabqkv, nullptr, 0, nullptr, qp, C, Mq, C, C, 0);
    // [kp|vp] = LN(kv) @ [Wk|Wv] + [bk|bv]         [32768, 1024] bf16 (dominant)
    gemm_bf16_kernel<<<dim3(1024 / 64, gy_kv), 256, 0, stream>>>(
        kvn, C, wXAqkv + C, 3 * C, xabqkv + C, nullptr, 0, nullptr, kvp, 1024, Mkv, 1024, C, 0);
    // masked cross-attention
    flash_attn_kernel<<<dim3(B * H, (Nq + 15) / 16), 32, 0, stream>>>(
        qp, C, kvp, 1024, kvp + 512, 1024, xaout, C, mask, ranyb, H, Nq, HW, scale);
    // q1 = q + attn @ Wo + bo                      [800, 512] f32
    gemm_bf16_kernel<<<dim3(C / 64, gy_q), 256, 0, stream>>>(
        xaout, C, wXAo, C, xabo, q_in, C, q1, nullptr, C, Mq, C, C, 0);

    layernorm_bf16_kernel<<<Mq, 256, 0, stream>>>(q1, n1g, n1b, qn1, C);
    // self-attn QKV                                [800, 1536] bf16
    gemm_bf16_kernel<<<dim3(1536 / 64, gy_q), 256, 0, stream>>>(
        qn1, C, wSAqkv, 1536, sabqkv, nullptr, 0, nullptr, saqkv, 1536, Mq, 1536, C, 0);
    flash_attn_kernel<<<dim3(B * H, (Nq + 15) / 16), 32, 0, stream>>>(
        saqkv, 1536, saqkv + 512, 1536, saqkv + 1024, 1536, saout, C,
        nullptr, nullptr, H, Nq, Nq, scale);
    // q2 = q1 + attn @ Wo + bo
    gemm_bf16_kernel<<<dim3(C / 64, gy_q), 256, 0, stream>>>(
        saout, C, wSAo, C, sabo, q1, C, q2, nullptr, C, Mq, C, C, 0);

    layernorm_bf16_kernel<<<Mq, 256, 0, stream>>>(q2, n2g, n2b, qn2, C);
    // h = relu(LN(q2) @ W1 + b1)                   [800, 2048] bf16
    gemm_bf16_kernel<<<dim3(F / 64, gy_q), 256, 0, stream>>>(
        qn2, C, wF1, F, ffb1, nullptr, 0, nullptr, hbuf, F, Mq, F, C, 1);
    // out = q2 + h @ W2 + b2                       [800, 512] f32 -> d_out
    gemm_bf16_kernel<<<dim3(C / 64, gy_q), 256, 0, stream>>>(
        hbuf, F, wF2, C, ffb2, q2, C, (float*)d_out, nullptr, C, Mq, C, F, 0);
}
